// mynet_30039001268550
// MI455X (gfx1250) — matrix operand order repaired and checked by reading, not compiled, not run on hardware
//
#include <hip/hip_runtime.h>
#include <cmath>

typedef __attribute__((ext_vector_type(16))) _Float16 v16h;
typedef __attribute__((ext_vector_type(8)))  float    v8f;

#define NODES 32
#define GRAPHS 4096
#define DEG 8
#define EPG 256                 // edges per graph = 32*8
#define FEAT 10
#define WAVES 4                 // graphs (waves) per workgroup
#define E_TOTAL (GRAPHS*NODES*DEG)

__device__ __forceinline__ float leaky(float v, float s) { return v > 0.f ? v : v * s; }

// ---- CDNA5 async global->LDS copy (ASYNCcnt-tracked), 16 bytes per active lane ----
__device__ __forceinline__ void async_g2l_b128(void* lds_ptr, const void* gptr) {
  asm volatile("global_load_async_to_lds_b128 %0, %1, off"
               :: "v"((unsigned)(uintptr_t)lds_ptr), "v"(gptr)
               : "memory");
}
__device__ __forceinline__ void wait_async0() {
  asm volatile("s_wait_asynccnt 0x0" ::: "memory");
}

__launch_bounds__(WAVES * 32)
__global__ void gatnet_fused_kernel(
    const float* __restrict__ x,
    const int*   __restrict__ edge_index,   // [2, E] (row0 = src, row1 = dst)
    const float* __restrict__ Wq, const float* __restrict__ Wk, const float* __restrict__ Wv,
    const float* __restrict__ W1, const float* __restrict__ a1s, const float* __restrict__ a1d, const float* __restrict__ b1,
    const float* __restrict__ W2, const float* __restrict__ a2s, const float* __restrict__ a2d, const float* __restrict__ b2,
    const float* __restrict__ W3, const float* __restrict__ a3s, const float* __restrict__ a3d, const float* __restrict__ b3,
    const float* __restrict__ W4, const float* __restrict__ a4s, const float* __restrict__ a4d, const float* __restrict__ b4,
    const float* __restrict__ fcW, const float* __restrict__ fcb,
    float* __restrict__ out)
{
  // Shared-per-WG weight staging (transposed: sW[col][k]) + per-wave buffers.
  __shared__ __align__(16) _Float16 sW[64][64];            // 8 KB
  __shared__ __align__(16) _Float16 sA[WAVES][NODES][64];  // 16 KB  (GEMM A, f16)
  __shared__ __align__(16) float    sH[WAVES][NODES][64];  // 32 KB  (GEMM D / scratch)
  __shared__ unsigned char sDst[WAVES][EPG];               // 1 KB
  __shared__ float sEs[WAVES][NODES][4];                   // 2 KB
  __shared__ float sEd[WAVES][NODES][4];                   // 2 KB
  __shared__ float sPool[WAVES][64];                       // 1 KB

  const int tid  = threadIdx.x;
  const int w    = tid >> 5;
  const int lane = tid & 31;
  const int g    = blockIdx.x * WAVES + w;

  // ---- stage this graph's edge destinations via async LDS copies ----
  // raw int32 dsts land in per-wave sH scratch, then get packed to uchar.
  {
    int* eScratch = (int*)&sH[w][0][0];                    // 1 KB of the 8 KB wave scratch
    const char* gbase = (const char*)(edge_index + E_TOTAL + g * EPG);
    for (int r = 0; r < 2; ++r) {                          // 2 x (32 lanes x 16B) = 1 KB
      int chunk = lane + 32 * r;
      async_g2l_b128((char*)eScratch + chunk * 16, gbase + chunk * 16);
    }
    wait_async0();
    __syncthreads();
    for (int j = 0; j < DEG; ++j) {
      int e = lane * DEG + j;
      sDst[w][e] = (unsigned char)(eScratch[e] & (NODES - 1));
    }
  }
  __syncthreads();

  // ---- per-graph self attention (VALU; att_in[s][i] = x[node i][s]) ----
  float* scratch = &sH[w][0][0];        // reuse sH before first GEMM
  float* xb = scratch;                  // [32][10]
  float* qb = scratch + 320;            // [10][32]
  float* kb = scratch + 640;            // [10][32]
  float* vb = scratch + 960;            // [10][32]
  float* al = scratch + 1280;           // [10][10]

  for (int s = 0; s < FEAT; ++s)
    xb[lane * FEAT + s] = x[(g * NODES + lane) * FEAT + s];
  __syncthreads();

  for (int s = 0; s < FEAT; ++s) {      // lane computes column `lane` of Q/K/V
    float q = 0.f, k = 0.f, v = 0.f;
    for (int kk = 0; kk < NODES; ++kk) {
      float xv = xb[kk * FEAT + s];
      q += xv * Wq[kk * NODES + lane];
      k += xv * Wk[kk * NODES + lane];
      v += xv * Wv[kk * NODES + lane];
    }
    qb[s * NODES + lane] = q;
    kb[s * NODES + lane] = k;
    vb[s * NODES + lane] = v;
  }
  __syncthreads();

  if (lane < FEAT) {                    // row-softmax of Q K^T
    int s = lane;
    float sc[FEAT];
    float m = -3.4e38f;
    for (int t = 0; t < FEAT; ++t) {
      float acc = 0.f;
      for (int i = 0; i < NODES; ++i) acc += qb[s * NODES + i] * kb[t * NODES + i];
      sc[t] = acc; m = fmaxf(m, acc);
    }
    float den = 0.f;
    for (int t = 0; t < FEAT; ++t) { sc[t] = __expf(sc[t] - m); den += sc[t]; }
    float inv = 1.f / den;
    for (int t = 0; t < FEAT; ++t) al[s * FEAT + t] = sc[t] * inv;
  }
  __syncthreads();

  float attout[FEAT];                   // out[s][lane] -> node `lane`, feature s
  for (int s = 0; s < FEAT; ++s) {
    float acc = 0.f;
    for (int t = 0; t < FEAT; ++t) acc += al[s * FEAT + t] * vb[t * NODES + lane];
    attout[s] = acc;
  }
  __syncthreads();                      // done with sH scratch
  for (int c = 0; c < 64; ++c)
    sA[w][lane][c] = (c < FEAT) ? (_Float16)attout[c] : (_Float16)0.f;

  const float* Ws[4] = {W1, W2, W3, W4};
  const float* As[4] = {a1s, a2s, a3s, a4s};
  const float* Ad[4] = {a1d, a2d, a3d, a4d};
  const float* Bs[4] = {b1, b2, b3, b4};

  float cur[64];                        // node `lane` features after each layer

#pragma unroll
  for (int l = 0; l < 4; ++l) {
    const int Ksteps = (l == 0) ? 1 : 2;        // K = 32 (zero-padded from 10) or 64
    const int H = (l < 3) ? 4 : 1;
    const int C = (l < 3) ? 16 : 64;
    const int Kdim = (l == 0) ? FEAT : 64;
    const float* Wl = Ws[l];

    __syncthreads();
    // ---- async-stage raw f32 weights into LDS (overlaid on sH block) ----
    float* wstage = &sH[0][0][0];               // up to 16 KB of the 32 KB sH block
    {
      const int nchunks = Kdim * 16;            // Kdim*64 floats / 4 per 16B chunk
      const char* gb = (const char*)Wl;
      char* lb = (char*)wstage;
      for (int c = tid; c < nchunks; c += WAVES * 32)
        async_g2l_b128(lb + c * 16, gb + c * 16);
      wait_async0();
    }
    __syncthreads();
    // convert/transpose: sW[col][k] in f16, zero-padded along K
    for (int idx = tid; idx < 64 * 64; idx += WAVES * 32) {
      int r = idx >> 6, c = idx & 63;           // r = k, c = out column
      float v = (r < Kdim) ? wstage[r * 64 + c] : 0.f;
      sW[c][r] = (_Float16)v;
    }
    __syncthreads();

    // ---- GEMM: sH[w] (32x64, f32) = sA[w] (32xK, f16) @ W (Kx64) via WMMA ----
    {
      const int rowInTile = lane & 15;
      const bool hiHalf = lane >= 16;
      for (int mt = 0; mt < 2; ++mt) {
        const int arow = mt * 16 + rowInTile;
        for (int nt = 0; nt < 4; ++nt) {
          const int ncol = nt * 16 + rowInTile;
          v8f acc = {0.f, 0.f, 0.f, 0.f, 0.f, 0.f, 0.f, 0.f};
          for (int ks = 0; ks < Ksteps; ++ks) {
            const int ka  = ks * 32 + (hiHalf ? 8 : 0);    // A: lanes<16 K0-7/16-23, hi K8-15/24-31
            const int kbB = ks * 32 + (hiHalf ? 8 : 0);    // B: lanes<16 K0-7/16-23, hi K8-15/24-31
            v16h a, b;
            for (int j = 0; j < 8; ++j) {
              a[j]     = sA[w][arow][ka + j];
              a[j + 8] = sA[w][arow][ka + 16 + j];
            }
            for (int j = 0; j < 8; ++j) {
              b[j]     = sW[ncol][kbB + j];
              b[j + 8] = sW[ncol][kbB + 16 + j];
            }
            acc = __builtin_amdgcn_wmma_f32_16x16x32_f16(
                false, a, false, b, (short)0, acc, false, false);
          }
          const int drowBase = mt * 16 + (hiHalf ? 8 : 0); // D: VGPR r -> row r (+8 hi)
          for (int r = 0; r < 8; ++r)
            sH[w][drowBase + r][nt * 16 + rowInTile] = acc[r];
        }
      }
    }
    __syncthreads();

    // ---- per-node attention logits e_s, e_d ----
    {
      const float* as = As[l];
      const float* ad = Ad[l];
      for (int hh = 0; hh < H; ++hh) {
        float ss = 0.f, dd = 0.f;
        for (int c = 0; c < C; ++c) {
          float hv = sH[w][lane][hh * C + c];
          ss += hv * as[hh * C + c];
          dd += hv * ad[hh * C + c];
        }
        sEs[w][lane][hh] = ss;
        sEd[w][lane][hh] = dd;
      }
    }
    __syncthreads();

    // ---- edge softmax + aggregation: lane = dst node t (atomic-free) ----
    {
      const int t = lane;
      float mh[4], den[4], evSelf[4], outv[64];
      for (int hh = 0; hh < H; ++hh) {
        evSelf[hh] = leaky(sEs[w][t][hh] + sEd[w][t][hh], 0.2f);
        mh[hh] = evSelf[hh];
      }
      for (int e = 0; e < EPG; ++e) {
        if (sDst[w][e] == t) {
          int s = e >> 3;
          for (int hh = 0; hh < H; ++hh)
            mh[hh] = fmaxf(mh[hh], leaky(sEs[w][s][hh] + sEd[w][t][hh], 0.2f));
        }
      }
      for (int hh = 0; hh < H; ++hh) den[hh] = 0.f;
      for (int c = 0; c < 64; ++c) outv[c] = 0.f;
      for (int hh = 0; hh < H; ++hh) {                 // self loop (src = t)
        float wv = __expf(evSelf[hh] - mh[hh]);
        den[hh] += wv;
        for (int c = 0; c < C; ++c) outv[hh * C + c] += wv * sH[w][t][hh * C + c];
      }
      for (int e = 0; e < EPG; ++e) {
        if (sDst[w][e] == t) {
          int s = e >> 3;
          for (int hh = 0; hh < H; ++hh) {
            float ev = leaky(sEs[w][s][hh] + sEd[w][t][hh], 0.2f);
            float wv = __expf(ev - mh[hh]);
            den[hh] += wv;
            for (int c = 0; c < C; ++c) outv[hh * C + c] += wv * sH[w][s][hh * C + c];
          }
        }
      }
      const float* bb = Bs[l];
      for (int hh = 0; hh < H; ++hh) {
        float inv = 1.f / (den[hh] + 1e-16f);
        for (int c = 0; c < C; ++c) {
          float v = outv[hh * C + c] * inv + bb[hh * C + c];
          cur[hh * C + c] = leaky(v, 0.01f);
        }
      }
    }
    __syncthreads();                                    // everyone done reading sH/sA
    for (int c = 0; c < 64; ++c) sA[w][lane][c] = (_Float16)cur[c];
  }

  // ---- mean pool + FC(64->2) + softmax ----
  __syncthreads();
  for (int c = 0; c < 64; ++c) sH[w][lane][c] = cur[c];
  __syncthreads();
  for (int j = 0; j < 2; ++j) {
    int c = lane * 2 + j;
    float s = 0.f;
    for (int t = 0; t < NODES; ++t) s += sH[w][t][c];
    sPool[w][c] = s * (1.f / NODES);
  }
  __syncthreads();
  if (lane == 0) {
    float lg[2];
    for (int k2 = 0; k2 < 2; ++k2) {
      float acc = fcb[k2];
      for (int c = 0; c < 64; ++c) acc += sPool[w][c] * fcW[c * 2 + k2];
      lg[k2] = acc;
    }
    float m = fmaxf(lg[0], lg[1]);
    float p0 = __expf(lg[0] - m), p1 = __expf(lg[1] - m);
    float inv = 1.f / (p0 + p1);
    out[g * 2 + 0] = lg[0];
    out[g * 2 + 1] = lg[1];
    out[2 * GRAPHS + g * 2 + 0] = p0 * inv;
    out[2 * GRAPHS + g * 2 + 1] = p1 * inv;
  }
}

extern "C" void kernel_launch(void* const* d_in, const int* in_sizes, int n_in,
                              void* d_out, int out_size, void* d_ws, size_t ws_size,
                              hipStream_t stream) {
  (void)in_sizes; (void)n_in; (void)out_size; (void)d_ws; (void)ws_size;
  const float* x   = (const float*)d_in[0];
  const int*   ei  = (const int*)d_in[1];
  // d_in[2] = batch (unused; derivable from node index)
  const float* Wq  = (const float*)d_in[3];
  const float* Wk  = (const float*)d_in[4];
  const float* Wv  = (const float*)d_in[5];
  const float* W1  = (const float*)d_in[6];
  const float* a1s = (const float*)d_in[7];
  const float* a1d = (const float*)d_in[8];
  const float* b1  = (const float*)d_in[9];
  const float* W2  = (const float*)d_in[10];
  const float* a2s = (const float*)d_in[11];
  const float* a2d = (const float*)d_in[12];
  const float* b2  = (const float*)d_in[13];
  const float* W3  = (const float*)d_in[14];
  const float* a3s = (const float*)d_in[15];
  const float* a3d = (const float*)d_in[16];
  const float* b3  = (const float*)d_in[17];
  const float* W4  = (const float*)d_in[18];
  const float* a4s = (const float*)d_in[19];
  const float* a4d = (const float*)d_in[20];
  const float* b4  = (const float*)d_in[21];
  const float* fcW = (const float*)d_in[22];
  const float* fcb = (const float*)d_in[23];
  float* out = (float*)d_out;

  dim3 grid(GRAPHS / WAVES);   // 1024 workgroups
  dim3 block(WAVES * 32);      // 128 threads = 4 wave32 = 4 graphs
  gatnet_fused_kernel<<<grid, block, 0, stream>>>(
      x, ei, Wq, Wk, Wv,
      W1, a1s, a1d, b1, W2, a2s, a2d, b2,
      W3, a3s, a3d, b3, W4, a4s, a4d, b4,
      fcW, fcb, out);
}
